// Attention_variants_33990371180695
// MI455X (gfx1250) — matrix operand order repaired and checked by reading, not compiled, not run on hardware
//
#include <hip/hip_runtime.h>
#include <hip/hip_fp16.h>

typedef _Float16 v16h __attribute__((ext_vector_type(16)));
typedef _Float16 v8h  __attribute__((ext_vector_type(8)));
typedef float    v8f  __attribute__((ext_vector_type(8)));

union V16U { v16h v; v8h h[2]; };

// Problem constants (from reference)
constexpr int Vv = 4, Bc = 4, Nn = 4096, Cc = 768, Hh = 12, Dd = 64;
constexpr float SCALE = 0.125f; // D^-0.5

// Tiling
constexpr int MT = 16;                 // tokens per workgroup
constexpr int KT = 32;                 // WMMA K step (f16)
constexpr int TPB = 512;               // threads per block (16 wave32)
constexpr int WAVES = TPB / 32;
constexpr int NTILES = Cc / 16;        // 48 column tiles
constexpr int TPW = NTILES / WAVES;    // 3 column tiles per wave

constexpr int S_Q  = Cc * Cc;          // WqT elems
constexpr int S_KV = Cc * 2 * Cc;      // WkvT elems
constexpr int S_P  = Cc * Cc;          // WpT elems

// ---------------- weight transpose + f16 convert (prep) ----------------
__global__ void prep_weights(const float* __restrict__ Wq,
                             const float* __restrict__ Wkv,
                             const float* __restrict__ Wp,
                             _Float16* __restrict__ WqT,
                             _Float16* __restrict__ WkvT,
                             _Float16* __restrict__ WpT) {
    int i = blockIdx.x * blockDim.x + threadIdx.x;
    if (i < S_Q) {
        int n = i / Cc, k = i % Cc;
        WqT[i] = (_Float16)Wq[(size_t)k * Cc + n];
    } else if (i < S_Q + S_KV) {
        int j = i - S_Q;
        int n = j / Cc, k = j % Cc;
        WkvT[j] = (_Float16)Wkv[(size_t)k * (2 * Cc) + n];
    } else if (i < S_Q + S_KV + S_P) {
        int j = i - S_Q - S_KV;
        int n = j / Cc, k = j % Cc;
        WpT[j] = (_Float16)Wp[(size_t)k * Cc + n];
    }
}

// ---------------- fragment loaders ----------------
// B fragment (32x16 f16): lane n%16 = output column; lanes 0-15 K=kb..kb+15,
// lanes 16-31 K=kb+16..kb+31 (contiguous in transposed weight row).
__device__ __forceinline__ v16h bfrag(const _Float16* __restrict__ WT, int colBase,
                                      int kb, int lane) {
    const v8h* p = reinterpret_cast<const v8h*>(
        WT + (size_t)(colBase + (lane & 15)) * Cc + kb + 16 * (lane >> 4));
    V16U u; u.h[0] = p[0]; u.h[1] = p[1];
    return u.v;
}

// A fragment (16x32 f16) from LDS row-major [MT][Cc]:
// lanes 0-15: K kb..kb+15; lanes 16-31: K kb+16..kb+31
__device__ __forceinline__ v16h afrag_lds(const _Float16* base, int kb, int lane) {
    const v8h* p = reinterpret_cast<const v8h*>(
        base + (size_t)(lane & 15) * Cc + kb + 16 * (lane >> 4));
    V16U u; u.h[0] = p[0]; u.h[1] = p[1];
    return u.v;
}

#define WMMA_F16(A, B, C) \
    __builtin_amdgcn_wmma_f32_16x16x32_f16(false, (A), false, (B), (short)0, (C), false, false)

// ---------------- fused attention kernel ----------------
__global__ __launch_bounds__(TPB) void attn_variants_fused(
    const float* __restrict__ x,          // [B*N, C]
    const float* __restrict__ variants,   // [V, B*N, C]
    const _Float16* __restrict__ WqT,     // [C][C]   (n-major)
    const _Float16* __restrict__ WkvT,    // [2C][C]  (n-major)
    const _Float16* __restrict__ WpT,     // [C][C]   (n-major)
    const float* __restrict__ bp,         // [C]
    float* __restrict__ out)              // [B*N, C]
{
    __shared__ _Float16 lds_p[Vv][MT][Cc];     // 96 KB  variant inputs (f16)
    __shared__ _Float16 lds_q[MT][Cc];         // 24 KB  Q
    __shared__ _Float16 lds_xo[MT][Cc];        // 24 KB  x (phase A) then out_pre (D->E)
    __shared__ float    part[MT][NTILES][Vv];  // 12 KB  per-tile logit partials
    __shared__ float    attn_s[MT][Hh][Vv];    // 3 KB   softmax weights

    const int tid  = threadIdx.x;
    const int wave = tid >> 5;
    const int lane = tid & 31;
    const int hi   = lane >> 4;
    const int nn   = lane & 15;
    const int tok0 = blockIdx.x * MT;

    // ---- stage x + 4 variant tiles into LDS as f16 (coalesced float4 reads) ----
    {
        constexpr int NV4 = 5 * MT * Cc / 4;     // 15360 float4s (x + 4 variants)
        for (int i = tid; i < NV4; i += TPB) {
            int e = i * 4;
            int s = e / (MT * Cc);               // 0 = x, 1..4 = variants
            int r = (e / Cc) % MT;
            int c = e % Cc;
            const float* src = (s == 0)
                ? (x + (size_t)(tok0 + r) * Cc + c)
                : (variants + (size_t)(s - 1) * (Bc * Nn * Cc) + (size_t)(tok0 + r) * Cc + c);
            const float4 f = *reinterpret_cast<const float4*>(src);
            _Float16* dst = (s == 0) ? &lds_xo[r][c] : &lds_p[s - 1][r][c];
            dst[0] = (_Float16)f.x; dst[1] = (_Float16)f.y;
            dst[2] = (_Float16)f.z; dst[3] = (_Float16)f.w;
        }
    }
    __syncthreads();

    // ---- Phase A: Q = X @ Wq  (software-pipelined: prefetch next k-step) ----
    {
        v8f acc[TPW];
#pragma unroll
        for (int j = 0; j < TPW; ++j) acc[j] = (v8f){0.f,0.f,0.f,0.f,0.f,0.f,0.f,0.f};
        v16h a_cur = afrag_lds(&lds_xo[0][0], 0, lane);
        v16h b_cur[TPW];
#pragma unroll
        for (int j = 0; j < TPW; ++j) b_cur[j] = bfrag(WqT, (wave * TPW + j) * 16, 0, lane);
        for (int kb = 0; kb < Cc; kb += KT) {
            const int kn = kb + KT;
            v16h a_nxt = a_cur;
            v16h b_nxt[TPW];
#pragma unroll
            for (int j = 0; j < TPW; ++j) b_nxt[j] = b_cur[j];
            if (kn < Cc) {
                a_nxt = afrag_lds(&lds_xo[0][0], kn, lane);
#pragma unroll
                for (int j = 0; j < TPW; ++j)
                    b_nxt[j] = bfrag(WqT, (wave * TPW + j) * 16, kn, lane);
            }
#pragma unroll
            for (int j = 0; j < TPW; ++j) acc[j] = WMMA_F16(a_cur, b_cur[j], acc[j]);
            a_cur = a_nxt;
#pragma unroll
            for (int j = 0; j < TPW; ++j) b_cur[j] = b_nxt[j];
        }
#pragma unroll
        for (int j = 0; j < TPW; ++j) {
            int cb = (wave * TPW + j) * 16;
#pragma unroll
            for (int g = 0; g < 8; ++g)
                lds_q[g + 8 * hi][cb + nn] = (_Float16)acc[j][g];
        }
    }
    __syncthreads();   // Q visible to all waves' K epilogues

    // ---- Phase B: K GEMM, B shared over all 4 variants; logits fused in epilogue ----
#pragma unroll
    for (int j = 0; j < TPW; ++j) {
        const int ct = wave * TPW + j;
        const int cb = ct * 16;
        v8f acc[Vv];
#pragma unroll
        for (int v = 0; v < Vv; ++v) acc[v] = (v8f){0.f,0.f,0.f,0.f,0.f,0.f,0.f,0.f};
        v16h b_cur = bfrag(WkvT, cb, 0, lane);
        v16h a_cur[Vv];
#pragma unroll
        for (int v = 0; v < Vv; ++v) a_cur[v] = afrag_lds(&lds_p[v][0][0], 0, lane);
        for (int kb = 0; kb < Cc; kb += KT) {
            const int kn = kb + KT;
            v16h b_nxt = b_cur;
            v16h a_nxt[Vv];
#pragma unroll
            for (int v = 0; v < Vv; ++v) a_nxt[v] = a_cur[v];
            if (kn < Cc) {
                b_nxt = bfrag(WkvT, cb, kn, lane);
#pragma unroll
                for (int v = 0; v < Vv; ++v) a_nxt[v] = afrag_lds(&lds_p[v][0][0], kn, lane);
            }
#pragma unroll
            for (int v = 0; v < Vv; ++v) acc[v] = WMMA_F16(a_cur[v], b_cur, acc[v]);
            b_cur = b_nxt;
#pragma unroll
            for (int v = 0; v < Vv; ++v) a_cur[v] = a_nxt[v];
        }
        // epilogue: logit partials  part[t][ct][v] = sum_{col in tile} q[t][col]*k_v[t][col]
        float qv[8];
#pragma unroll
        for (int g = 0; g < 8; ++g)
            qv[g] = (float)lds_q[g + 8 * hi][cb + nn];
#pragma unroll
        for (int v = 0; v < Vv; ++v) {
#pragma unroll
            for (int g = 0; g < 8; ++g) {
                float p = qv[g] * acc[v][g];
                // reduce across the 16 lanes of this half (deterministic tree)
                p += __shfl_xor(p, 1, 16);
                p += __shfl_xor(p, 2, 16);
                p += __shfl_xor(p, 4, 16);
                p += __shfl_xor(p, 8, 16);
                if (nn == 0) part[g + 8 * hi][ct][v] = p;
            }
        }
    }
    __syncthreads();

    // ---- Phase C: sum tile partials per head, softmax over the 4 variants ----
    if (tid < MT * Hh) {
        int t = tid / Hh, h = tid % Hh;
        float lg[Vv];
#pragma unroll
        for (int v = 0; v < Vv; ++v)
            lg[v] = (part[t][h * 4 + 0][v] + part[t][h * 4 + 1][v] +
                     part[t][h * 4 + 2][v] + part[t][h * 4 + 3][v]) * SCALE;
        float m = fmaxf(fmaxf(lg[0], lg[1]), fmaxf(lg[2], lg[3]));
        float e0 = __expf(lg[0] - m), e1 = __expf(lg[1] - m);
        float e2 = __expf(lg[2] - m), e3 = __expf(lg[3] - m);
        float inv = 1.f / (e0 + e1 + e2 + e3);
        attn_s[t][h][0] = e0 * inv; attn_s[t][h][1] = e1 * inv;
        attn_s[t][h][2] = e2 * inv; attn_s[t][h][3] = e3 * inv;
    }
    __syncthreads();

    // ---- Phase D: V GEMM with attention-combine epilogue -> out_pre (reuse lds_xo) ----
#pragma unroll
    for (int j = 0; j < TPW; ++j) {
        int cb = (wave * TPW + j) * 16;
        v8f acc[Vv];
#pragma unroll
        for (int v = 0; v < Vv; ++v) acc[v] = (v8f){0.f,0.f,0.f,0.f,0.f,0.f,0.f,0.f};
        v16h b_cur = bfrag(WkvT, Cc + cb, 0, lane);
        v16h a_cur[Vv];
#pragma unroll
        for (int v = 0; v < Vv; ++v) a_cur[v] = afrag_lds(&lds_p[v][0][0], 0, lane);
        for (int kb = 0; kb < Cc; kb += KT) {
            const int kn = kb + KT;
            v16h b_nxt = b_cur;
            v16h a_nxt[Vv];
#pragma unroll
            for (int v = 0; v < Vv; ++v) a_nxt[v] = a_cur[v];
            if (kn < Cc) {
                b_nxt = bfrag(WkvT, Cc + cb, kn, lane);
#pragma unroll
                for (int v = 0; v < Vv; ++v) a_nxt[v] = afrag_lds(&lds_p[v][0][0], kn, lane);
            }
#pragma unroll
            for (int v = 0; v < Vv; ++v) acc[v] = WMMA_F16(a_cur[v], b_cur, acc[v]);
            b_cur = b_nxt;
#pragma unroll
            for (int v = 0; v < Vv; ++v) a_cur[v] = a_nxt[v];
        }
        int head = cb >> 6;  // 16-col tile lies within one head (64 cols)
#pragma unroll
        for (int g = 0; g < 8; ++g) {
            int t = g + 8 * hi;
            float o = attn_s[t][head][0] * acc[0][g]
                    + attn_s[t][head][1] * acc[1][g]
                    + attn_s[t][head][2] * acc[2][g]
                    + attn_s[t][head][3] * acc[3][g];
            lds_xo[t][cb + nn] = (_Float16)o;          // out_pre overlays x buffer
        }
    }
    __syncthreads();

    // ---- Phase E: OUT = out_pre @ Wp + bp  (software-pipelined) ----
    {
        v8f acc[TPW];
#pragma unroll
        for (int j = 0; j < TPW; ++j) acc[j] = (v8f){0.f,0.f,0.f,0.f,0.f,0.f,0.f,0.f};
        v16h a_cur = afrag_lds(&lds_xo[0][0], 0, lane);
        v16h b_cur[TPW];
#pragma unroll
        for (int j = 0; j < TPW; ++j) b_cur[j] = bfrag(WpT, (wave * TPW + j) * 16, 0, lane);
        for (int kb = 0; kb < Cc; kb += KT) {
            const int kn = kb + KT;
            v16h a_nxt = a_cur;
            v16h b_nxt[TPW];
#pragma unroll
            for (int j = 0; j < TPW; ++j) b_nxt[j] = b_cur[j];
            if (kn < Cc) {
                a_nxt = afrag_lds(&lds_xo[0][0], kn, lane);
#pragma unroll
                for (int j = 0; j < TPW; ++j)
                    b_nxt[j] = bfrag(WpT, (wave * TPW + j) * 16, kn, lane);
            }
#pragma unroll
            for (int j = 0; j < TPW; ++j) acc[j] = WMMA_F16(a_cur, b_cur[j], acc[j]);
            a_cur = a_nxt;
#pragma unroll
            for (int j = 0; j < TPW; ++j) b_cur[j] = b_nxt[j];
        }
#pragma unroll
        for (int j = 0; j < TPW; ++j) {
            int cb = (wave * TPW + j) * 16;
            float bias = bp[cb + nn];
#pragma unroll
            for (int g = 0; g < 8; ++g)
                out[(size_t)(tok0 + g + 8 * hi) * Cc + cb + nn] = acc[j][g] + bias;
        }
    }
}

// ---------------- host launcher ----------------
extern "C" void kernel_launch(void* const* d_in, const int* in_sizes, int n_in,
                              void* d_out, int out_size, void* d_ws, size_t ws_size,
                              hipStream_t stream) {
    const float* x        = (const float*)d_in[0];
    const float* variants = (const float*)d_in[1];
    // d_in[2] = num_layer (unused)
    const float* Wq  = (const float*)d_in[3];
    const float* Wkv = (const float*)d_in[4];
    const float* Wp  = (const float*)d_in[5];
    const float* bp  = (const float*)d_in[6];
    float* out = (float*)d_out;

    _Float16* WqT  = (_Float16*)d_ws;
    _Float16* WkvT = WqT + S_Q;
    _Float16* WpT  = WkvT + S_KV;

    constexpr int totW = S_Q + S_KV + S_P;
    prep_weights<<<(totW + 255) / 256, 256, 0, stream>>>(Wq, Wkv, Wp, WqT, WkvT, WpT);

    const int nTok = Bc * Nn;                 // 16384
    attn_variants_fused<<<nTok / MT, TPB, 0, stream>>>(x, variants, WqT, WkvT, WpT, bp, out);
}
